// MedicalVQAModel_53403623358686
// MI455X (gfx1250) — compile-verified, matrix operand order checked by reading, not run on hardware
//
#include <hip/hip_runtime.h>

#define IN_DIM 1536
#define OUT_DIM 768
#define N_EXPERTS 8
#define BATCH 16384
#define NSLOT (2 * BATCH)

typedef __attribute__((ext_vector_type(16))) __bf16 v16bf;
typedef __attribute__((ext_vector_type(8))) float v8f;
typedef __attribute__((ext_vector_type(4))) int v4i;

union Frag {
    v16bf v;
    uint4 q[2];
};

__device__ __forceinline__ unsigned short f2bf(float f) {
    unsigned int u = __float_as_uint(f);
    u += 0x7FFFu + ((u >> 16) & 1u);   // round-to-nearest-even
    return (unsigned short)(u >> 16);
}

#if __has_builtin(__builtin_amdgcn_global_load_async_to_lds_b128) && \
    __has_builtin(__builtin_amdgcn_s_wait_asynccnt)
#define USE_ASYNC_LDS 1
#define AS1 __attribute__((address_space(1)))
#define AS3 __attribute__((address_space(3)))
#endif

// ---------------------------------------------------------------- zero
__global__ void moe_zero_kernel(float* __restrict__ out,
                                int* __restrict__ counts,
                                int* __restrict__ cursors) {
    size_t i = ((size_t)blockIdx.x * blockDim.x + threadIdx.x) * 4;
    *(float4*)(out + i) = make_float4(0.f, 0.f, 0.f, 0.f);
    if (blockIdx.x == 0 && threadIdx.x < N_EXPERTS) {
        counts[threadIdx.x] = 0;
        cursors[threadIdx.x] = 0;
    }
}

// ------------------------------------------------------- W f32 -> bf16
__global__ void moe_convert_w_kernel(const float* __restrict__ w,
                                     unsigned short* __restrict__ wbf) {
    size_t i = ((size_t)blockIdx.x * blockDim.x + threadIdx.x) * 8;
    float4 f0 = *(const float4*)(w + i);
    float4 f1 = *(const float4*)(w + i + 4);
    uint4 o;
    o.x = (unsigned)f2bf(f0.x) | ((unsigned)f2bf(f0.y) << 16);
    o.y = (unsigned)f2bf(f0.z) | ((unsigned)f2bf(f0.w) << 16);
    o.z = (unsigned)f2bf(f1.x) | ((unsigned)f2bf(f1.y) << 16);
    o.w = (unsigned)f2bf(f1.z) | ((unsigned)f2bf(f1.w) << 16);
    *(uint4*)(wbf + i) = o;
}

// ---------------------------------------------------------------- router
// One wave32 per token. Fuses the x f32->bf16 conversion into the same
// pass over x (x is read exactly once from HBM).
__global__ void moe_router_kernel(const float* __restrict__ x,
                                  const float* __restrict__ rw,
                                  const float* __restrict__ rb,
                                  unsigned short* __restrict__ xbf,
                                  int* __restrict__ topk_idx,
                                  float* __restrict__ topk_w,
                                  int* __restrict__ counts) {
    const int wave = threadIdx.x >> 5;
    const int lane = threadIdx.x & 31;
    const int b = blockIdx.x * 8 + wave;
    const float* xr = x + (size_t)b * IN_DIM;
    unsigned short* xb = xbf + (size_t)b * IN_DIM;

    float acc[N_EXPERTS];
#pragma unroll
    for (int e = 0; e < N_EXPERTS; ++e) acc[e] = 0.f;

    for (int d = lane * 4; d < IN_DIM; d += 128) {
        float4 xv = *(const float4*)(xr + d);
        uint2 pk;
        pk.x = (unsigned)f2bf(xv.x) | ((unsigned)f2bf(xv.y) << 16);
        pk.y = (unsigned)f2bf(xv.z) | ((unsigned)f2bf(xv.w) << 16);
        *(uint2*)(xb + d) = pk;
#pragma unroll
        for (int e = 0; e < N_EXPERTS; ++e) {
            float4 wv = *(const float4*)(rw + (size_t)e * IN_DIM + d);
            acc[e] = fmaf(xv.x, wv.x, acc[e]);
            acc[e] = fmaf(xv.y, wv.y, acc[e]);
            acc[e] = fmaf(xv.z, wv.z, acc[e]);
            acc[e] = fmaf(xv.w, wv.w, acc[e]);
        }
    }
#pragma unroll
    for (int off = 16; off >= 1; off >>= 1) {
#pragma unroll
        for (int e = 0; e < N_EXPERTS; ++e)
            acc[e] += __shfl_xor(acc[e], off, 32);
    }
    if (lane == 0) {
        float l[N_EXPERTS];
#pragma unroll
        for (int e = 0; e < N_EXPERTS; ++e) l[e] = acc[e] + rb[e];
        int i1 = 0;
#pragma unroll
        for (int e = 1; e < N_EXPERTS; ++e) if (l[e] > l[i1]) i1 = e;
        int i2 = (i1 == 0) ? 1 : 0;
#pragma unroll
        for (int e = 0; e < N_EXPERTS; ++e)
            if (e != i1 && l[e] > l[i2]) i2 = e;
        // renormalized top-2 softmax weights (global softmax denominator cancels)
        float w1 = 1.f / (1.f + expf(l[i2] - l[i1]));
        topk_idx[2 * b] = i1;
        topk_idx[2 * b + 1] = i2;
        topk_w[2 * b] = w1;
        topk_w[2 * b + 1] = 1.f - w1;
        atomicAdd(&counts[i1], 1);
        atomicAdd(&counts[i2], 1);
    }
}

// ---------------------------------------------------------------- scan
__global__ void moe_scan_kernel(const int* __restrict__ counts,
                                int* __restrict__ offsets,
                                int* __restrict__ tok_slot,
                                float* __restrict__ w_slot) {
    if (threadIdx.x == 0) {
        int off = 0;
        for (int e = 0; e < N_EXPERTS; ++e) {
            offsets[e] = off;
            int c = counts[e];
            int pc = (c + 15) & ~15;               // pad to 16-token tiles
            for (int j = c; j < pc; ++j) {
                tok_slot[off + j] = 0;             // dummy token
                w_slot[off + j] = 0.f;             // zero contribution
            }
            off += pc;
        }
        offsets[N_EXPERTS] = off;
    }
}

// ---------------------------------------------------------------- build
__global__ void moe_build_kernel(const int* __restrict__ topk_idx,
                                 const float* __restrict__ topk_w,
                                 const int* __restrict__ offsets,
                                 int* __restrict__ cursors,
                                 int* __restrict__ tok_slot,
                                 float* __restrict__ w_slot) {
    int i = blockIdx.x * blockDim.x + threadIdx.x;   // 0 .. 2B-1
    int e = topk_idx[i];
    float w = topk_w[i];
    int pos = atomicAdd(&cursors[e], 1);
    int slot = offsets[e] + pos;
    tok_slot[slot] = i >> 1;
    w_slot[slot] = w;
}

// ------------------------------------------------------------ expert GEMM
// block = 256 threads = 8 waves; tile = 16 slots x 256 out cols
// wave w covers out cols [obase+w*32, obase+w*32+32): two 16x16 WMMA tiles
__launch_bounds__(256, 2)
__global__ void moe_expert_gemm_kernel(const unsigned short* __restrict__ xbf,
                                       const unsigned short* __restrict__ wbf,
                                       const float* __restrict__ eb,
                                       const int* __restrict__ counts,
                                       const int* __restrict__ offsets,
                                       const int* __restrict__ tok_slot,
                                       const float* __restrict__ w_slot,
                                       float* __restrict__ out) {
    const int e = blockIdx.z;
    const int tile = blockIdx.x;
    if (tile * 16 >= counts[e]) return;            // block-uniform: EXEC stays all-1s
    const int slot0 = offsets[e] + tile * 16;

    __shared__ unsigned short ldsA[16 * 128];      // 16 slots x 128-K bf16 tile
    __shared__ int sTok[16];
    __shared__ float sW[16];

    const int tid = threadIdx.x;
    const int wave = tid >> 5;
    const int lane = tid & 31;
    const int half = lane >> 4;                    // lane group (K split per ISA layout)
    const int n = lane & 15;                       // M row (A) / N col (B,C)

    if (tid < 16) sTok[tid] = tok_slot[slot0 + tid];
    else if (tid < 32) sW[tid - 16] = w_slot[slot0 + tid - 16];
    __syncthreads();

    const int obase = blockIdx.y * 256 + wave * 32;
    const unsigned short* w0 = wbf + ((size_t)e * OUT_DIM + obase + n) * IN_DIM;
    const unsigned short* w1 = w0 + (size_t)16 * IN_DIM;

    // staging coords: thread t copies 8 consecutive bf16 (16B) of row t/16
    const int srow = tid >> 4;
    const int scol = (tid & 15) * 8;
    const unsigned short* gsrc = xbf + (size_t)sTok[srow] * IN_DIM + scol;
    unsigned short* ldst = &ldsA[srow * 128 + scol];

    v8f c0 = {};
    v8f c1 = {};

    for (int kc = 0; kc < IN_DIM; kc += 128) {
        // ---- stage gathered bf16 x rows into LDS ----
        if (kc + 128 < IN_DIM)
            __builtin_prefetch((const void*)(gsrc + kc + 128), 0, 0); // global_prefetch_b8
#ifdef USE_ASYNC_LDS
        // async global->LDS (ASYNCcnt-tracked), no VGPR round-trip
        __builtin_amdgcn_global_load_async_to_lds_b128(
            (AS1 v4i*)(v4i*)(gsrc + kc), (AS3 v4i*)ldst, 0, 0);
        __builtin_amdgcn_s_wait_asynccnt(0);
#else
        *(uint4*)ldst = *(const uint4*)(gsrc + kc);
#endif
        __syncthreads();

        // ---- 4 K-steps of 32, two 16x16 N-tiles per wave ----
#pragma unroll
        for (int ks = 0; ks < 128; ks += 32) {
            Frag a, b0, b1;
            // A fragment, ISA 16-bit 16x32 layout:
            //   lanes 0-15: K = ks+0..7 / ks+16..23 ; lanes 16-31: +8
            const int abase = n * 128 + ks + half * 8;
            a.q[0] = *(const uint4*)&ldsA[abase];
            a.q[1] = *(const uint4*)&ldsA[abase + 16];
            const int kb = kc + ks + half * 8;
            b0.q[0] = *(const uint4*)(w0 + kb);
            b0.q[1] = *(const uint4*)(w0 + kb + 16);
            b1.q[0] = *(const uint4*)(w1 + kb);
            b1.q[1] = *(const uint4*)(w1 + kb + 16);
            c0 = __builtin_amdgcn_wmma_f32_16x16x32_bf16(
                false, a.v, false, b0.v, (short)0, c0, false, false);
            c1 = __builtin_amdgcn_wmma_f32_16x16x32_bf16(
                false, a.v, false, b1.v, (short)0, c1, false, false);
        }
        __syncthreads();
    }

    // ---- epilogue: bias + exact-erf GELU + routing weight, atomic combine ----
    const float bias0 = eb[(size_t)e * OUT_DIM + obase + n];
    const float bias1 = eb[(size_t)e * OUT_DIM + obase + 16 + n];
#pragma unroll
    for (int r = 0; r < 8; ++r) {
        const int row = r + half * 8;              // C/D layout: lanes 16-31 hold M+8
        const int t = sTok[row];
        const float wgt = sW[row];
        float* orow = out + (size_t)t * OUT_DIM + obase + n;
        float v0 = c0[r] + bias0;
        float g0 = 0.5f * v0 * (1.f + erff(v0 * 0.70710678118654752f));
        atomicAdd(orow, wgt * g0);
        float v1 = c1[r] + bias1;
        float g1 = 0.5f * v1 * (1.f + erff(v1 * 0.70710678118654752f));
        atomicAdd(orow + 16, wgt * g1);
    }
}

// ---------------------------------------------------------------- launch
extern "C" void kernel_launch(void* const* d_in, const int* in_sizes, int n_in,
                              void* d_out, int out_size, void* d_ws, size_t ws_size,
                              hipStream_t stream) {
    const float* x   = (const float*)d_in[0];
    const float* rw  = (const float*)d_in[1];
    const float* rb  = (const float*)d_in[2];
    const float* ew  = (const float*)d_in[3];
    const float* ebv = (const float*)d_in[4];
    float* out = (float*)d_out;

    char* ws = (char*)d_ws;
    size_t off = 0;
    auto take = [&](size_t bytes) -> void* {
        void* p = (void*)(ws + off);
        off += (bytes + 255) & ~(size_t)255;
        return p;
    };
    unsigned short* wbf = (unsigned short*)take((size_t)N_EXPERTS * OUT_DIM * IN_DIM * 2);
    unsigned short* xbf = (unsigned short*)take((size_t)BATCH * IN_DIM * 2);
    int*   topk_idx = (int*)take((size_t)NSLOT * 4);
    float* topk_w   = (float*)take((size_t)NSLOT * 4);
    int*   counts   = (int*)take(64);
    int*   offsets  = (int*)take(64);
    int*   cursors  = (int*)take(64);
    int*   tok_slot = (int*)take((size_t)(NSLOT + 256) * 4);
    float* w_slot   = (float*)take((size_t)(NSLOT + 256) * 4);

    moe_zero_kernel<<<dim3((BATCH * OUT_DIM) / 1024), 256, 0, stream>>>(out, counts, cursors);
    moe_convert_w_kernel<<<dim3((N_EXPERTS * OUT_DIM * IN_DIM) / 2048), 256, 0, stream>>>(ew, wbf);
    moe_router_kernel<<<dim3(BATCH / 8), 256, 0, stream>>>(x, rw, rb, xbf,
                                                           topk_idx, topk_w, counts);
    moe_scan_kernel<<<1, 32, 0, stream>>>(counts, offsets, tok_slot, w_slot);
    moe_build_kernel<<<dim3(NSLOT / 256), 256, 0, stream>>>(topk_idx, topk_w, offsets,
                                                            cursors, tok_slot, w_slot);
    moe_expert_gemm_kernel<<<dim3(BATCH / 16, OUT_DIM / 256, N_EXPERTS), 256, 0, stream>>>(
        xbf, wbf, ebv, counts, offsets, tok_slot, w_slot, out);
}
